// ExpFilter_77446850281783
// MI455X (gfx1250) — matrix operand order repaired and checked by reading, not compiled, not run on hardware
//
#include <hip/hip_runtime.h>

typedef _Float16 half_t;
typedef __attribute__((ext_vector_type(16))) _Float16 v16h;
typedef __attribute__((ext_vector_type(8)))  _Float16 v8h;
typedef __attribute__((ext_vector_type(8)))  float    v8f;
typedef __attribute__((ext_vector_type(4)))  unsigned int v4u;
typedef __attribute__((ext_vector_type(8)))  int v8i;
typedef __attribute__((ext_vector_type(4)))  int v4i;

#define T_DIM   2048
#define B_DIM   32
#define I_DIM   256
#define O_DIM   256
#define DECAY_F 0.36787944117144233f   // exp(-1)

#define NT      64      // output columns per workgroup (2 waves x 32)
#define CHUNK   32      // stored timesteps per workgroup
#define WARM    32      // warm-up recompute; decay^32 ~ 1.3e-14 << fp32 eps
#define SX      264     // padded LDS row stride (halves) for X tiles
#define SW      264     // padded LDS row stride (halves) for W tiles

#define LDS_W   (NT * SW)       // halves per W buffer
#define LDS_X   (B_DIM * SX)    // halves per X split buffer
#define XRAW_FLOATS (B_DIM * I_DIM)          // 8192 f32 = 32 KB
#define LDS_BYTES ((size_t)(2 * LDS_W + 2 * LDS_X) * 2 + (size_t)XRAW_FLOATS * 4)

union P4     { half_t h[4]; uint2 u; };
union F16x16 { v16h v; v8h h[2]; };

__device__ __forceinline__ void splitf(float x, half_t& hi, half_t& lo) {
  half_t h = (half_t)x;
  hi = h;
  lo = (half_t)(x - (float)h);
}

// TDM: DMA one contiguous 32KB X[t] tile (8192 f32) global -> LDS.
// D# group0: count=1, lds_addr, global_addr[56:0], type=2.
// D# group1: data_size=4B, tensor_dim0=tile_dim0=8192, dim1=1, stride0=8192.
__device__ __forceinline__ void tdm_load_x(const float* gsrc, unsigned lds_off) {
  unsigned long long ga = (unsigned long long)(size_t)gsrc;
  v4u g0;
  g0.x = 1u;                                        // count = 1 valid descriptor
  g0.y = lds_off;                                   // LDS byte address
  g0.z = (unsigned)(ga & 0xFFFFFFFFu);              // global_addr[31:0]
  g0.w = (unsigned)((ga >> 32) & 0x01FFFFFFu)       // global_addr[56:32]
       | 0x80000000u;                               // type = 2 ("image")
  v8i g1;
  g1[0] = 0x00020000;   // workgroup_mask=0, data_size=2 (4 bytes), no flags
  g1[1] = 0x20000000;   // tensor_dim0[15:0]=8192 in bits[31:16]
  g1[2] = 0x00010000;   // tensor_dim0[31:16]=0, tensor_dim1=1
  g1[3] = 0x20000000;   // tile_dim0 = 8192 in bits[31:16]
  g1[4] = 0x00000001;   // tile_dim1 = 1, tile_dim2 = 0
  g1[5] = 8192;         // tensor_dim0_stride[31:0]
  g1[6] = 0x20000000;   // stride0[47:32]=0, tensor_dim1_stride[15:0]=8192
  g1[7] = 0;            // tensor_dim1_stride[47:16]=0
  v4i z4 = {0, 0, 0, 0};                 // groups 2/3 unused (<=2D tensor)
  v8i z8 = {0, 0, 0, 0, 0, 0, 0, 0};     // 5th descriptor group (clang-23 form)
  __builtin_amdgcn_tensor_load_to_lds(g0, g1, z4, z4, z8, 0);
}

__global__ __launch_bounds__(64) void expfilter_fused(
    const float* __restrict__ Xg, const float* __restrict__ Wg,
    const float* __restrict__ Bg, float* __restrict__ Og)
{
  extern __shared__ __align__(16) char smem_raw[];
  half_t* Wh   = (half_t*)smem_raw;     // [NT][SW] f16 hi
  half_t* Wl   = Wh + LDS_W;            // [NT][SW] f16 lo
  half_t* Xh   = Wl + LDS_W;            // [B][SX]  f16 hi
  half_t* Xl   = Xh + LDS_X;            // [B][SX]  f16 lo
  float*  Xraw = (float*)(Xl + LDS_X);  // [B*I]    raw f32 (TDM destination)
  const unsigned xraw_lds = (unsigned)(size_t)Xraw;

  const int tid  = threadIdx.x;
  const int lane = tid & 31;
  const int wave = tid >> 5;        // 0..1, each wave owns a 32-wide N strip
  const int lo16 = lane & 15;
  const int hi16 = lane >> 4;       // 0 or 1

  const int n_blk   = blockIdx.x * NT;
  const int t0      = blockIdx.y * CHUNK;
  const int t_start = (t0 >= WARM) ? (t0 - WARM) : 0;
  const int t_end   = t0 + CHUNK;

  // Kick off async DMA of the first X tile; it overlaps the W staging below.
  if (wave == 0) tdm_load_x(Xg + (size_t)t_start * XRAW_FLOATS, xraw_lds);

  // ---- Stage W strip [n_blk .. n_blk+63] into LDS, split f16 hi/lo. One row/thread.
  {
    const float* wrow = Wg + (size_t)(n_blk + tid) * I_DIM;
    half_t* dh = Wh + tid * SW;
    half_t* dl = Wl + tid * SW;
    #pragma unroll 4
    for (int c = 0; c < I_DIM; c += 4) {
      float4 w4 = *(const float4*)(wrow + c);
      const float* wj = (const float*)&w4;
      P4 ph, pl;
      #pragma unroll
      for (int j = 0; j < 4; ++j) splitf(wj[j], ph.h[j], pl.h[j]);
      *(uint2*)(dh + c) = ph.u;
      *(uint2*)(dl + c) = pl.u;
    }
  }

  float bv[2];
  bv[0] = Bg[n_blk + wave * 32 + lo16];
  bv[1] = Bg[n_blk + wave * 32 + 16 + lo16];

  // IIR state lives in registers (WMMA C/D layout) across timesteps.
  v8f s[2][2] = {};

  if (wave == 0) __builtin_amdgcn_s_wait_tensorcnt((short)0);
  __syncthreads();   // first X tile + W strip visible to all waves

  for (int t = t_start; t < t_end; ++t) {
    // ---- Convert raw f32 tile (TDM-landed) -> split f16 hi/lo LDS buffers.
    #pragma unroll
    for (int i = 0; i < (XRAW_FLOATS / 4) / 64; ++i) {
      int idx = tid + i * 64;
      int row = idx >> 6;          // /64 float4 per row
      int c   = (idx & 63) << 2;
      float4 x4 = *(const float4*)(Xraw + (idx << 2));
      const float* xj = (const float*)&x4;
      P4 ph, pl;
      #pragma unroll
      for (int j = 0; j < 4; ++j) splitf(xj[j], ph.h[j], pl.h[j]);
      *(uint2*)(Xh + row * SX + c) = ph.u;
      *(uint2*)(Xl + row * SX + c) = pl.u;
    }
    __syncthreads();   // split buffers ready; raw buffer fully consumed

    // Refill raw buffer for t+1 via TDM; DMA overlaps the WMMA loop below.
    if (t + 1 < t_end && wave == 0)
      tdm_load_x(Xg + (size_t)(t + 1) * XRAW_FLOATS, xraw_lds);

    // ---- GEMM: y[32 x 64] = X[t] (32x256) * W^T (256x64), split-f16 3-product.
    v8f acc[2][2] = {};
    #pragma unroll
    for (int ks = 0; ks < I_DIM; ks += 32) {
      F16x16 ah[2], al[2], bh[2], bl[2];
      #pragma unroll
      for (int mt = 0; mt < 2; ++mt) {
        // A 16x32 f16 layout: lane -> M, elems 0..7 -> K=ks+8*hi16.., 8..15 -> +16
        int ra = (mt * 16 + lo16) * SX + ks + 8 * hi16;
        ah[mt].h[0] = *(const v8h*)(Xh + ra);
        ah[mt].h[1] = *(const v8h*)(Xh + ra + 16);
        al[mt].h[0] = *(const v8h*)(Xl + ra);
        al[mt].h[1] = *(const v8h*)(Xl + ra + 16);
      }
      #pragma unroll
      for (int nt = 0; nt < 2; ++nt) {
        // B 32x16 f16 layout: lane -> N, elems 0..7 -> K=ks+8*hi16.., 8..15 -> +16
        int rb = (wave * 32 + nt * 16 + lo16) * SW + ks + 8 * hi16;
        bh[nt].h[0] = *(const v8h*)(Wh + rb);
        bh[nt].h[1] = *(const v8h*)(Wh + rb + 16);
        bl[nt].h[0] = *(const v8h*)(Wl + rb);
        bl[nt].h[1] = *(const v8h*)(Wl + rb + 16);
      }
      #pragma unroll
      for (int mt = 0; mt < 2; ++mt) {
        #pragma unroll
        for (int nt = 0; nt < 2; ++nt) {
          acc[mt][nt] = __builtin_amdgcn_wmma_f32_16x16x32_f16(
              false, ah[mt].v, false, bh[nt].v, (short)0, acc[mt][nt], false, false);
          acc[mt][nt] = __builtin_amdgcn_wmma_f32_16x16x32_f16(
              false, ah[mt].v, false, bl[nt].v, (short)0, acc[mt][nt], false, false);
          acc[mt][nt] = __builtin_amdgcn_wmma_f32_16x16x32_f16(
              false, al[mt].v, false, bh[nt].v, (short)0, acc[mt][nt], false, false);
        }
      }
    }

    // ---- Bias + IIR update: s = (acc + b) + decay * s  (in C/D register layout)
    #pragma unroll
    for (int mt = 0; mt < 2; ++mt) {
      #pragma unroll
      for (int nt = 0; nt < 2; ++nt) {
        v8f y = acc[mt][nt] + bv[nt];
        s[mt][nt] = y + DECAY_F * s[mt][nt];
      }
    }

    // ---- Store (only inside this block's owned chunk)
    if (t >= t0) {
      float* orow = Og + (size_t)t * (B_DIM * O_DIM);
      #pragma unroll
      for (int mt = 0; mt < 2; ++mt) {
        #pragma unroll
        for (int nt = 0; nt < 2; ++nt) {
          int ncol = n_blk + wave * 32 + nt * 16 + lo16;
          #pragma unroll
          for (int v = 0; v < 8; ++v) {
            int m = mt * 16 + v + 8 * hi16;   // C/D layout: lanes16-31 hold M=8+v
            orow[(size_t)m * O_DIM + ncol] = s[mt][nt][v];
          }
        }
      }
    }

    // Raw tile for t+1 must be fully landed before next convert pass.
    if (wave == 0) __builtin_amdgcn_s_wait_tensorcnt((short)0);
    __syncthreads();
  }
}

extern "C" void kernel_launch(void* const* d_in, const int* in_sizes, int n_in,
                              void* d_out, int out_size, void* d_ws, size_t ws_size,
                              hipStream_t stream) {
  const float* X  = (const float*)d_in[0];   // [T, B, I]
  const float* W  = (const float*)d_in[1];   // [O, I]
  const float* bb = (const float*)d_in[2];   // [O]
  float* out = (float*)d_out;                // [T, B, O]

  dim3 grid(O_DIM / NT, T_DIM / CHUNK);      // (4, 64) = 256 workgroups
  dim3 block(64);                            // 2 waves (wave32)
  expfilter_fused<<<grid, block, LDS_BYTES, stream>>>(X, W, bb, out);

  (void)in_sizes; (void)n_in; (void)out_size; (void)d_ws; (void)ws_size;
}
